// DotReluBlock_13451837571889
// MI455X (gfx1250) — matrix operand order repaired and checked by reading, not compiled, not run on hardware
//
#include <hip/hip_runtime.h>
#include <hip/hip_bf16.h>

// ---------------------------------------------------------------------------
// MI455X (gfx1250) attention-with-query-softmax, bf16 WMMA pipeline.
// B=4, N=4096, H=128. Compute-bound: ~36 GFLOP vs ~0.5 GB traffic.
// Math path: v_wmma_f32_16x16x32_bf16 (wave32, fp32 accum).
// ---------------------------------------------------------------------------

typedef __bf16 bf16_t;
typedef __attribute__((ext_vector_type(16))) __bf16 bf16x16;
typedef __attribute__((ext_vector_type(8)))  __bf16 bf16x8;
typedef __attribute__((ext_vector_type(8)))  float  f32x8;
typedef __attribute__((ext_vector_type(8)))  float  v8f;

#define NB   4        // batch
#define NN   4096     // sequence
#define NH   128      // hidden

static constexpr float SCORE_SCALE = 0.0883883476483184405f; // 1/sqrt(128)

// workspace layout (bytes)
static constexpr size_t OFF_Q  = 0;                        // bf16 [B*N, H]   4 MiB
static constexpr size_t OFF_KT = 4ull  << 20;              // bf16 [B][H][N]  4 MiB
static constexpr size_t OFF_VF = 8ull  << 20;              // f32  [B*N, H]   8 MiB
static constexpr size_t OFF_VP = 16ull << 20;              // bf16 [B*N, H]   4 MiB (V/Z)
static constexpr size_t OFF_M  = 20ull << 20;              // f32  [B*N]      64 KiB
static constexpr size_t OFF_Z  = (20ull << 20) + (1ull << 16);
static constexpr size_t OFF_O  = 21ull << 20;              // f32  [B*N, H]   8 MiB
static constexpr size_t OFF_S  = 32ull << 20;              // bf16 [B][N][N]  128 MiB

__device__ inline v8f zero_v8f() {
  v8f z;
#pragma unroll
  for (int i = 0; i < 8; ++i) z[i] = 0.0f;
  return z;
}

__device__ inline v8f wmma_bf16x32(bf16x16 a, bf16x16 b, v8f c) {
  // (neg_a, A, neg_b, B, c_mod, C, reuse_a, reuse_b)
  return __builtin_amdgcn_wmma_f32_16x16x32_bf16(false, a, false, b, (short)0, c,
                                                 false, false);
}

// A fragment (16x32, bf16 row-major source). ISA layout: lane l (l16=l&15) holds
// row M=l16; in-lane element e -> K = kk + ((e>>3)<<4) + (l>>4)*8 + (e&7):
// two contiguous 8-element (16B) runs at kk+hi*8 and kk+16+hi*8.
__device__ inline bf16x16 load_a_bf16(const bf16_t* base, int ld, int row, int kk,
                                      int hi) {
  const bf16_t* p = base + (size_t)row * ld + kk + hi * 8;
  bf16x8 lo = *(const bf16x8*)(p);
  bf16x8 hh = *(const bf16x8*)(p + 16);
  bf16x16 a;
#pragma unroll
  for (int i = 0; i < 8; ++i) { a[i] = lo[i]; a[8 + i] = hh[i]; }
  return a;
}

// Same A layout but from fp32 source, converting to bf16 on the fly.
__device__ inline bf16x16 load_a_f32(const float* base, int ld, int row, int kk,
                                     int hi) {
  const float* p = base + (size_t)row * ld + kk + hi * 8;
  f32x8 lo = *(const f32x8*)(p);
  f32x8 hh = *(const f32x8*)(p + 16);
  bf16x16 a;
#pragma unroll
  for (int i = 0; i < 8; ++i) { a[i] = (bf16_t)lo[i]; a[8 + i] = (bf16_t)hh[i]; }
  return a;
}

// B fragment (32x16 KxN): lane l holds column N=col0+(l&15); element e -> K = kk + ((e>>3)<<4) + (l>>4)*8 + (e&7).
__device__ inline bf16x16 load_b_bf16(const bf16_t* base, int ld, int krow, int col0) {
  const int l = krow & 31, kk = krow - l;
  const bf16_t* p = base + (size_t)(kk + (l >> 4) * 8) * ld + col0 + (l & 15);
  bf16x16 b;
#pragma unroll
  for (int i = 0; i < 8; ++i) { b[i] = p[(size_t)i * ld]; b[8 + i] = p[(size_t)(16 + i) * ld]; }
  return b;
}

__device__ inline bf16x16 load_b_f32(const float* base, int ld, int krow, int col0) {
  const int l = krow & 31, kk = krow - l;
  const float* p = base + (size_t)(kk + (l >> 4) * 8) * ld + col0 + (l & 15);
  bf16x16 b;
#pragma unroll
  for (int i = 0; i < 8; ++i) { b[i] = (bf16_t)p[(size_t)i * ld]; b[8 + i] = (bf16_t)p[(size_t)(16 + i) * ld]; }
  return b;
}

// ---------------------------------------------------------------------------
// K1: projections. blockIdx.y: 0 -> Q=v@Wq+bq (bf16 row-major)
//                              1 -> K=k@Wk+bk (bf16, stored TRANSPOSED [H][N])
//                              2 -> V=k@Wv+bv (fp32 row-major)
// WG = 256 thr = 8 waves (4 row-groups x 2 col-groups); wave tile 32x64.
// ---------------------------------------------------------------------------
__global__ __launch_bounds__(256) void proj_kernel(
    const float* __restrict__ k_in, const float* __restrict__ v_in,
    const float* __restrict__ Wq, const float* __restrict__ bq,
    const float* __restrict__ Wk, const float* __restrict__ bk,
    const float* __restrict__ Wv, const float* __restrict__ bv,
    bf16_t* __restrict__ Qbf, bf16_t* __restrict__ KT, float* __restrict__ Vf) {
  const int pid = blockIdx.y;
  const float* src  = (pid == 0) ? v_in : k_in;
  const float* W    = (pid == 0) ? Wq : (pid == 1) ? Wk : Wv;
  const float* bias = (pid == 0) ? bq : (pid == 1) ? bk : bv;

  const int tid = threadIdx.x;
  const int l = tid & 31, l16 = l & 15, hi = l >> 4;
  const int w = tid >> 5, wr = w & 3, wc = w >> 2;
  const int rowBase = blockIdx.x * 128 + wr * 32;  // global row in [0, B*N)
  const int colBase = wc * 64;

  v8f acc[2][4];
#pragma unroll
  for (int mt = 0; mt < 2; ++mt)
#pragma unroll
    for (int nt = 0; nt < 4; ++nt) acc[mt][nt] = zero_v8f();

#pragma unroll
  for (int kk = 0; kk < NH; kk += 32) {
    bf16x16 afrag[2], bfrag[4];
#pragma unroll
    for (int mt = 0; mt < 2; ++mt)
      afrag[mt] = load_a_f32(src, NH, rowBase + mt * 16 + l16, kk, hi);
#pragma unroll
    for (int nt = 0; nt < 4; ++nt)
      bfrag[nt] = load_b_f32(W, NH, kk + l, colBase + nt * 16);
#pragma unroll
    for (int mt = 0; mt < 2; ++mt)
#pragma unroll
      for (int nt = 0; nt < 4; ++nt)
        acc[mt][nt] = wmma_bf16x32(afrag[mt], bfrag[nt], acc[mt][nt]);
  }

#pragma unroll
  for (int mt = 0; mt < 2; ++mt)
#pragma unroll
    for (int nt = 0; nt < 4; ++nt) {
      const int col = colBase + nt * 16 + l16;
      const float bia = bias[col];
#pragma unroll
      for (int r = 0; r < 8; ++r) {
        const int row = rowBase + mt * 16 + hi * 8 + r;  // D: lanes16-31 -> M+8
        const float val = acc[mt][nt][r] + bia;
        if (pid == 0) {
          Qbf[(size_t)row * NH + col] = (bf16_t)val;
        } else if (pid == 1) {
          const int b = row >> 12, n = row & (NN - 1);
          KT[((size_t)b * NH + col) * NN + n] = (bf16_t)val;
        } else {
          Vf[(size_t)row * NH + col] = val;
        }
      }
    }
}

// ---------------------------------------------------------------------------
// K2: S[b] = (Q[b] @ K[b]^T) / sqrt(H), bf16 out. 128x128 tile per WG.
// ---------------------------------------------------------------------------
__global__ __launch_bounds__(256) void scores_kernel(
    const bf16_t* __restrict__ Qbf, const bf16_t* __restrict__ KT,
    bf16_t* __restrict__ S) {
  const int b = blockIdx.z;
  const bf16_t* Qb  = Qbf + (size_t)b * NN * NH;
  const bf16_t* KTb = KT  + (size_t)b * NH * NN;
  bf16_t* Sb = S + (size_t)b * NN * NN;

  const int tid = threadIdx.x;
  const int l = tid & 31, l16 = l & 15, hi = l >> 4;
  const int w = tid >> 5, wr = w & 3, wc = w >> 2;
  const int rowBase = blockIdx.y * 128 + wr * 32;            // q
  const int colBase = blockIdx.x * 128 + wc * 64;            // k

  v8f acc[2][4];
#pragma unroll
  for (int mt = 0; mt < 2; ++mt)
#pragma unroll
    for (int nt = 0; nt < 4; ++nt) acc[mt][nt] = zero_v8f();

#pragma unroll
  for (int kk = 0; kk < NH; kk += 32) {
    if (kk + 32 < NH)
      __builtin_prefetch(KTb + (size_t)(kk + 32 + l) * NN + colBase, 0, 1);
    bf16x16 afrag[2], bfrag[4];
#pragma unroll
    for (int mt = 0; mt < 2; ++mt)
      afrag[mt] = load_a_bf16(Qb, NH, rowBase + mt * 16 + l16, kk, hi);
#pragma unroll
    for (int nt = 0; nt < 4; ++nt)
      bfrag[nt] = load_b_bf16(KTb, NN, kk + l, colBase + nt * 16);
#pragma unroll
    for (int mt = 0; mt < 2; ++mt)
#pragma unroll
      for (int nt = 0; nt < 4; ++nt)
        acc[mt][nt] = wmma_bf16x32(afrag[mt], bfrag[nt], acc[mt][nt]);
  }

#pragma unroll
  for (int mt = 0; mt < 2; ++mt)
#pragma unroll
    for (int nt = 0; nt < 4; ++nt) {
      const int col = colBase + nt * 16 + l16;
#pragma unroll
      for (int r = 0; r < 8; ++r) {
        const int row = rowBase + mt * 16 + hi * 8 + r;
        Sb[(size_t)row * NN + col] = (bf16_t)(acc[mt][nt][r] * SCORE_SCALE);
      }
    }
}

// ---------------------------------------------------------------------------
// K3: per key-column stats: m_k = max_q S, Z_k = sum_q exp(S - m_k).
// Also writes Vp[k,:] = V[k,:] / Z_k in bf16 (folds normalizer into V).
// ---------------------------------------------------------------------------
__global__ __launch_bounds__(256) void stats_kernel(
    const bf16_t* __restrict__ S, const float* __restrict__ Vf,
    float* __restrict__ mOut, float* __restrict__ zOut, bf16_t* __restrict__ Vp) {
  const int b = blockIdx.y;
  const int kcol = blockIdx.x * 256 + threadIdx.x;
  const bf16_t* Sb = S + (size_t)b * NN * NN;

  float m = -__builtin_inff();
  for (int q = 0; q < NN; ++q)
    m = fmaxf(m, (float)Sb[(size_t)q * NN + kcol]);
  float z = 0.0f;
  for (int q = 0; q < NN; ++q)
    z += __expf((float)Sb[(size_t)q * NN + kcol] - m);

  mOut[b * NN + kcol] = m;
  zOut[b * NN + kcol] = z;

  const float inv = 1.0f / z;
  const float* vr = Vf + ((size_t)b * NN + kcol) * NH;
  bf16_t* vp = Vp + ((size_t)b * NN + kcol) * NH;
#pragma unroll 4
  for (int h = 0; h < NH; ++h) vp[h] = (bf16_t)(vr[h] * inv);
}

// ---------------------------------------------------------------------------
// K4: P = exp(S - m_k) in place (bf16).
// ---------------------------------------------------------------------------
__global__ __launch_bounds__(256) void pexp_kernel(bf16_t* __restrict__ S,
                                                   const float* __restrict__ mArr) {
  const size_t idx = (size_t)blockIdx.x * 256 + threadIdx.x;  // < B*N*N
  const int b = (int)(idx >> 24);                             // N*N = 2^24
  const int kcol = (int)(idx & (NN - 1));
  const float m = mArr[b * NN + kcol];
  S[idx] = (bf16_t)__expf((float)S[idx] - m);
}

// ---------------------------------------------------------------------------
// K5: O[b] = P[b] (N x N bf16) @ Vp[b] (N x H bf16), fp32 out.
// WG covers 128 rows x all 128 cols; K loop over N in steps of 32.
// ---------------------------------------------------------------------------
__global__ __launch_bounds__(256) void out_gemm_kernel(
    const bf16_t* __restrict__ P, const bf16_t* __restrict__ Vp,
    float* __restrict__ O) {
  const int b = blockIdx.y;
  const bf16_t* Pb  = P  + (size_t)b * NN * NN;
  const bf16_t* Vpb = Vp + (size_t)b * NN * NH;
  float* Ob = O + (size_t)b * NN * NH;

  const int tid = threadIdx.x;
  const int l = tid & 31, l16 = l & 15, hi = l >> 4;
  const int w = tid >> 5, wr = w & 3, wc = w >> 2;
  const int rowBase = blockIdx.x * 128 + wr * 32;  // q
  const int colBase = wc * 64;                     // h

  v8f acc[2][4];
#pragma unroll
  for (int mt = 0; mt < 2; ++mt)
#pragma unroll
    for (int nt = 0; nt < 4; ++nt) acc[mt][nt] = zero_v8f();

  for (int kk = 0; kk < NN; kk += 32) {
    if (kk + 64 < NN) {
      __builtin_prefetch(Vpb + (size_t)(kk + 64 + l) * NH + colBase, 0, 1);
      __builtin_prefetch(Pb + (size_t)(rowBase + l16) * NN + kk + 64, 0, 1);
    }
    bf16x16 afrag[2], bfrag[4];
#pragma unroll
    for (int mt = 0; mt < 2; ++mt)
      afrag[mt] = load_a_bf16(Pb, NN, rowBase + mt * 16 + l16, kk, hi);
#pragma unroll
    for (int nt = 0; nt < 4; ++nt)
      bfrag[nt] = load_b_bf16(Vpb, NH, kk + l, colBase + nt * 16);
#pragma unroll
    for (int mt = 0; mt < 2; ++mt)
#pragma unroll
      for (int nt = 0; nt < 4; ++nt)
        acc[mt][nt] = wmma_bf16x32(afrag[mt], bfrag[nt], acc[mt][nt]);
  }

#pragma unroll
  for (int mt = 0; mt < 2; ++mt)
#pragma unroll
    for (int nt = 0; nt < 4; ++nt) {
      const int col = colBase + nt * 16 + l16;
#pragma unroll
      for (int r = 0; r < 8; ++r) {
        const int row = rowBase + mt * 16 + hi * 8 + r;
        Ob[(size_t)row * NH + col] = acc[mt][nt][r];
      }
    }
}

// ---------------------------------------------------------------------------
// K6: out[row] = O[row,:] . Wp + bp  (one wave32 per row, shuffle reduction)
// ---------------------------------------------------------------------------
__global__ __launch_bounds__(256) void final_kernel(
    const float* __restrict__ O, const float* __restrict__ Wp,
    const float* __restrict__ bp, float* __restrict__ out) {
  const int tid = threadIdx.x;
  const int l = tid & 31, w = tid >> 5;
  const int row = blockIdx.x * 8 + w;  // [0, B*N)
  const float* Orow = O + (size_t)row * NH;
  float s = 0.0f;
#pragma unroll
  for (int i = 0; i < 4; ++i) s += Orow[l + 32 * i] * Wp[l + 32 * i];
#pragma unroll
  for (int off = 16; off > 0; off >>= 1) s += __shfl_xor(s, off, 32);
  if (l == 0) out[row] = s + bp[0];
}

// ---------------------------------------------------------------------------
extern "C" void kernel_launch(void* const* d_in, const int* in_sizes, int n_in,
                              void* d_out, int out_size, void* d_ws, size_t ws_size,
                              hipStream_t stream) {
  (void)in_sizes; (void)n_in; (void)out_size; (void)ws_size;
  const float* k_in = (const float*)d_in[0];
  const float* v_in = (const float*)d_in[1];
  const float* Wq = (const float*)d_in[2];
  const float* bq = (const float*)d_in[3];
  const float* Wk = (const float*)d_in[4];
  const float* bk = (const float*)d_in[5];
  const float* Wv = (const float*)d_in[6];
  const float* bv = (const float*)d_in[7];
  const float* Wp = (const float*)d_in[8];
  const float* bp = (const float*)d_in[9];
  float* out = (float*)d_out;

  char* ws = (char*)d_ws;
  bf16_t* Qbf = (bf16_t*)(ws + OFF_Q);
  bf16_t* KT  = (bf16_t*)(ws + OFF_KT);
  float*  Vf  = (float*)(ws + OFF_VF);
  bf16_t* Vp  = (bf16_t*)(ws + OFF_VP);
  float*  Mv  = (float*)(ws + OFF_M);
  float*  Zv  = (float*)(ws + OFF_Z);
  float*  O   = (float*)(ws + OFF_O);
  bf16_t* S   = (bf16_t*)(ws + OFF_S);

  proj_kernel<<<dim3(NB * NN / 128, 3), 256, 0, stream>>>(
      k_in, v_in, Wq, bq, Wk, bk, Wv, bv, Qbf, KT, Vf);
  scores_kernel<<<dim3(NN / 128, NN / 128, NB), 256, 0, stream>>>(Qbf, KT, S);
  stats_kernel<<<dim3(NN / 256, NB), 256, 0, stream>>>(S, Vf, Mv, Zv, Vp);
  pexp_kernel<<<dim3((unsigned)((size_t)NB * NN * NN / 256)), 256, 0, stream>>>(S, Mv);
  out_gemm_kernel<<<dim3(NN / 128, NB), 256, 0, stream>>>(S, Vp, O);
  final_kernel<<<dim3(NB * NN / 8), 256, 0, stream>>>(O, Wp, bp, out);
}
